// SelfAttentionConv_32710470926821
// MI455X (gfx1250) — matrix operand order repaired and checked by reading, not compiled, not run on hardware
//
#include <hip/hip_runtime.h>

// Problem constants (match reference)
constexpr int Bc = 8;
constexpr int Tc = 2048;
constexpr int Dc = 64;    // head dim == K
constexpr int Hc = 8;     // heads
constexpr int CH = Dc * Hc;   // 512 conv out channels
constexpr int KS = 5;
constexpr int KIC = Dc * KS;  // 320 im2col K

typedef __attribute__((ext_vector_type(16))) _Float16 v16h;
typedef __attribute__((ext_vector_type(8)))  float    v8f;

union ABFrag {
    v16h v;
    struct { int4 lo; int4 hi; } q;
};

// ---------------- conversion / repack kernels ----------------

__global__ void cvt_f32_to_f16(const float* __restrict__ src, _Float16* __restrict__ dst, int n) {
    int i = blockIdx.x * 256 + threadIdx.x;
    if (i < n) dst[i] = (_Float16)src[i];
}

// w: [512][64][5] -> wh: [512][320] with k = j*64 + i  (n-major, k-contiguous for B fragments)
__global__ void repack_conv_w(const float* __restrict__ w, _Float16* __restrict__ wh) {
    int idx = blockIdx.x * 256 + threadIdx.x;
    if (idx < CH * KIC) {
        int c = idx / KIC, k = idx % KIC;
        int j = k >> 6, i = k & 63;
        wh[idx] = (_Float16)w[(c * Dc + i) * KS + j];
    }
}

// ---------------- conv (im2col GEMM) kernel ----------------

template <bool GUARD>
__device__ inline v8f conv_mac(const _Float16* __restrict__ xb,   // xh + b*T*64
                               int t_a,
                               const _Float16* __restrict__ wb,   // wgt + (c0+row)*KIC
                               int hi) {
    v8f c = {};
#pragma unroll
    for (int ks = 0; ks < 10; ++ks) {      // k = ks*32..+31 ; j = ks/2, i = (ks&1)*32..
        const int j  = ks >> 1;
        const int ib = (ks & 1) * 32;
        const int tp = t_a - 4 + j;        // causal left pad = 4
        ABFrag af;
        if (!GUARD || tp >= 0) {
            const _Float16* p = xb + (size_t)tp * Dc + ib + hi * 8;
            af.q.lo = *reinterpret_cast<const int4*>(p);
            af.q.hi = *reinterpret_cast<const int4*>(p + 16);
        } else {
            int4 z = {0, 0, 0, 0};
            af.q.lo = z; af.q.hi = z;
        }
        ABFrag bf;
        const _Float16* wp = wb + ks * 32 + hi * 8;
        bf.q.lo = *reinterpret_cast<const int4*>(wp);
        bf.q.hi = *reinterpret_cast<const int4*>(wp + 16);
        c = __builtin_amdgcn_wmma_f32_16x16x32_f16(false, af.v, false, bf.v,
                                                   (short)0, c, false, false);
    }
    return c;
}

// grid: (B*T/16, CH/16, 3)  block: 32  (one wave, one 16x16 C tile)
__global__ __launch_bounds__(32)
void conv_qkv_kernel(const _Float16* __restrict__ xh,
                     const _Float16* __restrict__ wq, const _Float16* __restrict__ wk,
                     const _Float16* __restrict__ wv,
                     const float* __restrict__ bq, const float* __restrict__ bk,
                     _Float16* __restrict__ Qo, _Float16* __restrict__ Ko,
                     _Float16* __restrict__ Vo) {
    const int lane = threadIdx.x;
    const int row  = lane & 15;
    const int hi   = lane >> 4;
    const int R0   = blockIdx.x * 16;      // global row over B*T (tiles never cross batch)
    const int c0   = blockIdx.y * 16;      // channel tile
    const int which = blockIdx.z;          // 0=q 1=k 2=v
    const int b  = R0 / Tc;
    const int t0 = R0 % Tc;
    const _Float16* wgt = (which == 0) ? wq : ((which == 1) ? wk : wv);

    const _Float16* xb = xh + (size_t)b * Tc * Dc;
    const _Float16* wb = wgt + (size_t)(c0 + row) * KIC;
    const int t_a = t0 + row;

    v8f c;
    if (t0 >= 4) {                          // uniform fast path: no tap can be negative
        c = conv_mac<false>(xb, t_a, wb, hi);
    } else {
        c = conv_mac<true>(xb, t_a, wb, hi);
    }

#pragma unroll
    for (int g = 0; g < 8; ++g) {
        const int M  = g + 8 * hi;
        const int t  = t0 + M;
        const int ch = c0 + row;
        const int d  = ch >> 3;            // channel = d*H + h (head fastest)
        const int h  = ch & 7;
        float val = c[g];
        if (which == 0) {
            val = (val + bq[ch]) * 0.125f;  // fold 1/sqrt(D) softmax scale into Q
            Qo[(((size_t)b * Hc + h) * Tc + t) * Dc + d] = (_Float16)val;
        } else if (which == 1) {
            val = val + bk[ch];
            Ko[(((size_t)b * Hc + h) * Tc + t) * Dc + d] = (_Float16)val;
        } else {
            Vo[(((size_t)b * Hc + h) * Dc + d) * Tc + t] = (_Float16)val;  // transposed [d][t]
        }
    }
}

// ---------------- flash attention kernel ----------------
// grid: (T/16, H, B)  block: 32.  One wave owns a 16-row Q tile, streams 64-key tiles.
__global__ __launch_bounds__(32)
void attn_kernel(const _Float16* __restrict__ Q, const _Float16* __restrict__ Km,
                 const _Float16* __restrict__ Vt, _Float16* __restrict__ O) {
    __shared__ __align__(16) _Float16 plds[16 * 72];  // 16x64 P staging, 72-half row stride

    const int lane = threadIdx.x;
    const int row  = lane & 15;
    const int hi   = lane >> 4;
    const int qb   = blockIdx.x * 16;
    const int h    = blockIdx.y;
    const int b    = blockIdx.z;

    const _Float16* qbase = Q  + (((size_t)b * Hc + h) * Tc + qb) * Dc; // [t][d]
    const _Float16* kbase = Km + (((size_t)b * Hc + h) * Tc) * Dc;      // [s][d]
    const _Float16* vbase = Vt + (((size_t)b * Hc + h) * Dc) * Tc;      // [d][t]

    // resident Q A-fragments (two 32-wide d-steps)
    v16h aq[2];
#pragma unroll
    for (int s = 0; s < 2; ++s) {
        ABFrag f;
        const _Float16* p = qbase + (size_t)row * Dc + s * 32 + hi * 8;
        f.q.lo = *reinterpret_cast<const int4*>(p);
        f.q.hi = *reinterpret_cast<const int4*>(p + 16);
        aq[s] = f.v;
    }

    v8f acc[4]; // O tiles over d (4 x 16)
#pragma unroll
    for (int nt = 0; nt < 4; ++nt) acc[nt] = {};
    float m[8], l[8];
#pragma unroll
    for (int g = 0; g < 8; ++g) { m[g] = -1.0e30f; l[g] = 0.0f; }

    const int send = qb + 16;
    for (int sb = 0; sb < send; sb += 64) {
        // prefetch next 64-key K/V tiles (global_prefetch_b8; no LOADcnt cost)
        if (sb + 64 < send) {
            __builtin_prefetch(kbase + (size_t)(sb + 64 + row * 4) * Dc, 0, 1);
            __builtin_prefetch(vbase + (size_t)(row * 4) * Tc + sb + 64, 0, 1);
        }

        // S = Q * K^T  (four 16x16 tiles over 64 keys, 2 d-steps each)
        v8f S[4];
#pragma unroll
        for (int st = 0; st < 4; ++st) {
            v8f c = {};
#pragma unroll
            for (int ds = 0; ds < 2; ++ds) {
                ABFrag bf;
                const _Float16* p =
                    kbase + (size_t)(sb + st * 16 + row) * Dc + ds * 32 + hi * 8;
                bf.q.lo = *reinterpret_cast<const int4*>(p);
                bf.q.hi = *reinterpret_cast<const int4*>(p + 16);
                c = __builtin_amdgcn_wmma_f32_16x16x32_f16(false, aq[ds], false, bf.v,
                                                           (short)0, c, false, false);
            }
            S[st] = c;
        }
        // causal mask on diagonal-overlapping tiles
        if (sb + 64 > qb) {
#pragma unroll
            for (int st = 0; st < 4; ++st)
#pragma unroll
                for (int g = 0; g < 8; ++g) {
                    const int qg = qb + g + 8 * hi;
                    const int sg = sb + st * 16 + row;
                    if (sg > qg) S[st][g] = -1.0e30f;
                }
        }
        // online softmax; rows live across the 16-lane half-groups
#pragma unroll
        for (int g = 0; g < 8; ++g) {
            float t0 = fmaxf(fmaxf(S[0][g], S[1][g]), fmaxf(S[2][g], S[3][g]));
            t0 = fmaxf(t0, __shfl_xor(t0, 1, 32));
            t0 = fmaxf(t0, __shfl_xor(t0, 2, 32));
            t0 = fmaxf(t0, __shfl_xor(t0, 4, 32));
            t0 = fmaxf(t0, __shfl_xor(t0, 8, 32));
            const float mn = fmaxf(m[g], t0);
            float p[4];
            float rs = 0.0f;
#pragma unroll
            for (int st = 0; st < 4; ++st) {
                p[st] = __expf(S[st][g] - mn);
                rs += p[st];
            }
            rs += __shfl_xor(rs, 1, 32);
            rs += __shfl_xor(rs, 2, 32);
            rs += __shfl_xor(rs, 4, 32);
            rs += __shfl_xor(rs, 8, 32);
            const float alpha = __expf(m[g] - mn);
            l[g] = l[g] * alpha + rs;
            m[g] = mn;
#pragma unroll
            for (int nt = 0; nt < 4; ++nt) acc[nt][g] *= alpha;
            const int M = g + 8 * hi;
#pragma unroll
            for (int st = 0; st < 4; ++st)
                plds[M * 72 + st * 16 + row] = (_Float16)p[st];
        }
        __syncthreads();
        // reload P (16x64) in A-fragment layout: two 32-wide k-steps
        v16h ap[2];
#pragma unroll
        for (int ds = 0; ds < 2; ++ds) {
            ABFrag pf;
            const _Float16* p = &plds[row * 72 + ds * 32 + hi * 8];
            pf.q.lo = *reinterpret_cast<const int4*>(p);
            pf.q.hi = *reinterpret_cast<const int4*>(p + 16);
            ap[ds] = pf.v;
        }
        __syncthreads();
        // O += P * V  (V transposed -> contiguous B-fragment loads)
#pragma unroll
        for (int ds = 0; ds < 2; ++ds) {
#pragma unroll
            for (int nt = 0; nt < 4; ++nt) {
                ABFrag bv;
                const _Float16* p =
                    vbase + (size_t)(nt * 16 + row) * Tc + sb + ds * 32 + hi * 8;
                bv.q.lo = *reinterpret_cast<const int4*>(p);
                bv.q.hi = *reinterpret_cast<const int4*>(p + 16);
                acc[nt] = __builtin_amdgcn_wmma_f32_16x16x32_f16(false, ap[ds], false, bv.v,
                                                                 (short)0, acc[nt], false, false);
            }
        }
    }
    // epilogue: O / l -> half, layout [b][t][h*64+d]
    _Float16* obase = O + ((size_t)b * Tc + qb) * CH + h * Dc;
#pragma unroll
    for (int g = 0; g < 8; ++g) {
        const float inv = 1.0f / l[g];
        const int M = g + 8 * hi;
#pragma unroll
        for (int nt = 0; nt < 4; ++nt)
            obase[(size_t)M * CH + nt * 16 + row] = (_Float16)(acc[nt][g] * inv);
    }
}

// ---------------- output projection kernel ----------------
// Y[BT,64] = X[BT,512] @ Wu^T + bu ; grid (B*T/16, 4), block 32
__global__ __launch_bounds__(32)
void proj_kernel(const _Float16* __restrict__ X, const _Float16* __restrict__ Wu,
                 const float* __restrict__ bu, float* __restrict__ Y) {
    const int lane = threadIdx.x;
    const int row  = lane & 15;
    const int hi   = lane >> 4;
    const size_t R0 = (size_t)blockIdx.x * 16;
    const int n0 = blockIdx.y * 16;
    v8f c = {};
#pragma unroll
    for (int ks = 0; ks < 16; ++ks) {
        ABFrag af;
        const _Float16* p = X + (R0 + row) * CH + ks * 32 + hi * 8;
        af.q.lo = *reinterpret_cast<const int4*>(p);
        af.q.hi = *reinterpret_cast<const int4*>(p + 16);
        ABFrag bf;
        const _Float16* q = Wu + (size_t)(n0 + row) * CH + ks * 32 + hi * 8;
        bf.q.lo = *reinterpret_cast<const int4*>(q);
        bf.q.hi = *reinterpret_cast<const int4*>(q + 16);
        c = __builtin_amdgcn_wmma_f32_16x16x32_f16(false, af.v, false, bf.v,
                                                   (short)0, c, false, false);
    }
#pragma unroll
    for (int g = 0; g < 8; ++g) {
        const int M = g + 8 * hi;
        Y[(R0 + M) * Dc + n0 + row] = c[g] + bu[n0 + row];
    }
}

// ---------------- host side ----------------

extern "C" void kernel_launch(void* const* d_in, const int* in_sizes, int n_in,
                              void* d_out, int out_size, void* d_ws, size_t ws_size,
                              hipStream_t stream) {
    const float* x  = (const float*)d_in[0];
    const float* wq = (const float*)d_in[1];
    const float* bq = (const float*)d_in[2];
    const float* wk = (const float*)d_in[3];
    const float* bk = (const float*)d_in[4];
    const float* wv = (const float*)d_in[5];
    const float* wu = (const float*)d_in[6];
    const float* bu = (const float*)d_in[7];
    float* out = (float*)d_out;

    constexpr size_t SZ_XH  = (size_t)Bc * Tc * Dc * 2;        // 2 MiB
    constexpr size_t SZ_QKV = (size_t)Bc * Hc * Tc * Dc * 2;   // 16 MiB each
    constexpr size_t SZ_W   = (size_t)CH * KIC * 2;            // 320 KiB each
    constexpr size_t SZ_ATT = (size_t)Bc * Tc * CH * 2;        // 16 MiB

    char* ws = (char*)d_ws;
    _Float16* xh   = (_Float16*)(ws);
    _Float16* qbuf = (_Float16*)(ws + SZ_XH);
    _Float16* kbuf = (_Float16*)(ws + SZ_XH + SZ_QKV);
    _Float16* vbuf = (_Float16*)(ws + SZ_XH + 2 * SZ_QKV);
    _Float16* att  = (_Float16*)(ws + SZ_XH + 3 * SZ_QKV);
    _Float16* wqh  = (_Float16*)(ws + SZ_XH + 3 * SZ_QKV + SZ_ATT);
    _Float16* wkh  = (_Float16*)(ws + SZ_XH + 3 * SZ_QKV + SZ_ATT + SZ_W);
    _Float16* wvh  = (_Float16*)(ws + SZ_XH + 3 * SZ_QKV + SZ_ATT + 2 * SZ_W);
    _Float16* wuh  = (_Float16*)(ws + SZ_XH + 3 * SZ_QKV + SZ_ATT + 3 * SZ_W);

    const int nX = Bc * Tc * Dc;
    cvt_f32_to_f16<<<(nX + 255) / 256, 256, 0, stream>>>(x, xh, nX);
    repack_conv_w<<<(CH * KIC + 255) / 256, 256, 0, stream>>>(wq, wqh);
    repack_conv_w<<<(CH * KIC + 255) / 256, 256, 0, stream>>>(wk, wkh);
    repack_conv_w<<<(CH * KIC + 255) / 256, 256, 0, stream>>>(wv, wvh);
    cvt_f32_to_f16<<<(Dc * CH + 255) / 256, 256, 0, stream>>>(wu, wuh, Dc * CH);

    conv_qkv_kernel<<<dim3(Bc * Tc / 16, CH / 16, 3), 32, 0, stream>>>(
        xh, wqh, wkh, wvh, bq, bk, qbuf, kbuf, vbuf);

    attn_kernel<<<dim3(Tc / 16, Hc, Bc), 32, 0, stream>>>(qbuf, kbuf, vbuf, att);

    proj_kernel<<<dim3(Bc * Tc / 16, Dc / 16), 32, 0, stream>>>(att, wuh, bu, out);
}
